// DecoderLSTM_26242250179284
// MI455X (gfx1250) — compile-verified, matrix operand order checked by reading, not run on hardware
//
#include <hip/hip_runtime.h>
#include <math.h>

#define B_ 32
#define N_ 64
#define F_ 1280
#define H_ 512
#define E_ 512
#define V_ 10000
#define T_ 80
#define SOS_ 1

typedef __attribute__((ext_vector_type(8)))  unsigned short ushort8;
typedef __attribute__((ext_vector_type(16))) __bf16         bf16x16;
typedef __attribute__((ext_vector_type(8)))  float          floatx8;

__device__ __forceinline__ unsigned short f2bf(float f) {
    union { float f; unsigned u; } v; v.f = f;
    unsigned r = v.u + 0x7FFFu + ((v.u >> 16) & 1u);   // round-to-nearest-even
    return (unsigned short)(r >> 16);
}

__device__ __forceinline__ float sigmf(float x) { return 1.0f / (1.0f + __expf(-x)); }

// Load a 16-element bf16 fragment for one lane.
// Per CDNA5 ISA 7.12.2 (16-bit A 16x32): lanes 0-15 hold K={kb+0..7, kb+16..23},
// lanes 16-31 hold K={kb+8..15, kb+24..31}.  koff = 8*(lane>=16).
// B fragments use the mirrored layout on a [N][K] (pre-transposed) matrix.
__device__ __forceinline__ bf16x16 load_frag(const unsigned short* p, int koff) {
    union { ushort8 h[2]; bf16x16 v; } u;
    u.h[0] = *(const ushort8*)(p + koff);
    u.h[1] = *(const ushort8*)(p + 16 + koff);
    return u.v;
}

// ---------------------------------------------------------------------------
// Wave-per-16x64-tile bf16 WMMA GEMM:
//   C[M,N] = act( A1[M,K1] @ Bt1[N,K1]^T (+ A2[M,K2] @ Bt2[N,K2]^T) + bias[N] )
// Bt matrices are pre-transposed ([N][K], bf16).  One wave32 owns a 16(M)x64(N)
// tile: 4 independent f32 accumulators -> 4 independent v_wmma dependency
// chains per wave (fills the matrix pipe at low occupancy) and each A fragment
// is reused for 4 WMMAs (1.25 fragment loads per matrix op instead of 2).
// Ragged N (N%64 != 0): invalid column-tiles read column 0 and skip the store.
// act: 0 = none, 1 = sigmoid.
// ---------------------------------------------------------------------------
__global__ __launch_bounds__(128) void wmma_gemm(
        const unsigned short* __restrict__ A1, int K1,
        const unsigned short* __restrict__ A2, int K2,
        const unsigned short* __restrict__ Bt1,
        const unsigned short* __restrict__ Bt2,
        const float* __restrict__ bias,
        float* __restrict__ C, int ldc,
        unsigned short* __restrict__ Cbf,
        int M, int N, int act) {
    int wave = threadIdx.x >> 5;
    int lane = threadIdx.x & 31;
    int ntn  = N >> 4;                       // 16-wide column tiles
    int ntg  = (ntn + 3) >> 2;               // groups of 4 column tiles
    int group = blockIdx.x * (blockDim.x >> 5) + wave;
    int total = (M >> 4) * ntg;
    if (group >= total) return;
    int mt  = group / ntg;
    int nt0 = (group % ntg) << 2;

    int half = lane >> 4;                    // 0 or 1
    int lm   = lane & 15;
    int koff = half << 3;

    bool valid[4];
    int  col[4];
#pragma unroll
    for (int j = 0; j < 4; ++j) {
        valid[j] = (nt0 + j) < ntn;
        col[j]   = valid[j] ? ((nt0 + j) * 16 + lm) : lm;   // invalid -> col 0 (safe)
    }

    floatx8 acc0 = {}, acc1 = {}, acc2 = {}, acc3 = {};

    {
        const unsigned short* arow = A1 + (size_t)(mt * 16 + lm) * K1;
        const unsigned short* b0 = Bt1 + (size_t)col[0] * K1;
        const unsigned short* b1 = Bt1 + (size_t)col[1] * K1;
        const unsigned short* b2 = Bt1 + (size_t)col[2] * K1;
        const unsigned short* b3 = Bt1 + (size_t)col[3] * K1;
        for (int kb = 0; kb < K1; kb += 32) {
            bf16x16 a = load_frag(arow + kb, koff);
            bf16x16 f0 = load_frag(b0 + kb, koff);
            bf16x16 f1 = load_frag(b1 + kb, koff);
            bf16x16 f2 = load_frag(b2 + kb, koff);
            bf16x16 f3 = load_frag(b3 + kb, koff);
            acc0 = __builtin_amdgcn_wmma_f32_16x16x32_bf16(false, a, false, f0, (short)0, acc0, false, false);
            acc1 = __builtin_amdgcn_wmma_f32_16x16x32_bf16(false, a, false, f1, (short)0, acc1, false, false);
            acc2 = __builtin_amdgcn_wmma_f32_16x16x32_bf16(false, a, false, f2, (short)0, acc2, false, false);
            acc3 = __builtin_amdgcn_wmma_f32_16x16x32_bf16(false, a, false, f3, (short)0, acc3, false, false);
        }
    }
    if (A2) {
        const unsigned short* arow = A2 + (size_t)(mt * 16 + lm) * K2;
        const unsigned short* b0 = Bt2 + (size_t)col[0] * K2;
        const unsigned short* b1 = Bt2 + (size_t)col[1] * K2;
        const unsigned short* b2 = Bt2 + (size_t)col[2] * K2;
        const unsigned short* b3 = Bt2 + (size_t)col[3] * K2;
        for (int kb = 0; kb < K2; kb += 32) {
            bf16x16 a = load_frag(arow + kb, koff);
            bf16x16 f0 = load_frag(b0 + kb, koff);
            bf16x16 f1 = load_frag(b1 + kb, koff);
            bf16x16 f2 = load_frag(b2 + kb, koff);
            bf16x16 f3 = load_frag(b3 + kb, koff);
            acc0 = __builtin_amdgcn_wmma_f32_16x16x32_bf16(false, a, false, f0, (short)0, acc0, false, false);
            acc1 = __builtin_amdgcn_wmma_f32_16x16x32_bf16(false, a, false, f1, (short)0, acc1, false, false);
            acc2 = __builtin_amdgcn_wmma_f32_16x16x32_bf16(false, a, false, f2, (short)0, acc2, false, false);
            acc3 = __builtin_amdgcn_wmma_f32_16x16x32_bf16(false, a, false, f3, (short)0, acc3, false, false);
        }
    }

    // C/D layout: VGPR r, lanes 0-15 -> M=r, lanes 16-31 -> M=8+r; N = lane%16.
    int mbase = mt * 16 + (half << 3);
    floatx8* accs[4] = { &acc0, &acc1, &acc2, &acc3 };
#pragma unroll
    for (int j = 0; j < 4; ++j) {
        if (!valid[j]) continue;
        int nn = (nt0 + j) * 16 + lm;
        float bv = bias ? bias[nn] : 0.0f;
        floatx8 a = *accs[j];
#pragma unroll
        for (int r = 0; r < 8; ++r) {
            float v = a[r] + bv;
            if (act == 1) v = sigmf(v);
            int m = mbase + r;
            C[(size_t)m * ldc + nn] = v;
            if (Cbf) Cbf[(size_t)m * N + nn] = f2bf(v);
        }
    }
}

// ---------------------------------------------------------------------------
// One-time conversion kernels
// ---------------------------------------------------------------------------
__global__ void cvt_kernel(const float* __restrict__ src,
                           unsigned short* __restrict__ dst, int count) {
    int i = blockIdx.x * blockDim.x + threadIdx.x;
    if (i < count) dst[i] = f2bf(src[i]);
}

// src: [K][N] f32 -> dst: [N][K] bf16
__global__ void transpose_cvt(const float* __restrict__ src,
                              unsigned short* __restrict__ dst, int K, int N) {
    int i = blockIdx.x * blockDim.x + threadIdx.x;
    if (i >= K * N) return;
    int k = i / N, n = i % N;
    dst[(size_t)n * K + k] = f2bf(src[i]);
}

__global__ void mean_kernel(const float* __restrict__ feat,
                            float* __restrict__ mean_enc,
                            unsigned short* __restrict__ mean_bf) {
    int b = blockIdx.x;
    for (int f = threadIdx.x; f < F_; f += blockDim.x) {
        float s = 0.0f;
        for (int n = 0; n < N_; ++n) s += feat[((size_t)b * N_ + n) * F_ + f];
        s *= (1.0f / (float)N_);
        mean_enc[(size_t)b * F_ + f] = s;
        mean_bf[(size_t)b * F_ + f] = f2bf(s);
    }
}

__global__ void init_x(const float* __restrict__ emb, float* __restrict__ x) {
    int b = blockIdx.x;
    for (int e = threadIdx.x; e < E_; e += blockDim.x)
        x[(size_t)b * E_ + e] = emb[(size_t)SOS_ * E_ + e];
}

// ---------------------------------------------------------------------------
// Attention: scores = relu(hq + enc_proj) @ Va + bVa, softmax over N,
// context = feat^T w, context *= gate (gate precomputed with sigmoid GEMM),
// writes attention weights to output and builds bf16 inp = [x | context].
// One block per batch element.
// ---------------------------------------------------------------------------
__global__ __launch_bounds__(256) void attention_kernel(
        const float* __restrict__ hq, const float* __restrict__ enc_proj,
        const float* __restrict__ Va, const float* __restrict__ bVa,
        const float* __restrict__ feat, const float* __restrict__ gate,
        const float* __restrict__ x,
        float* __restrict__ attn_out, int attn_ld,
        unsigned short* __restrict__ inp_bf) {
    int b = blockIdx.x, tid = threadIdx.x;
    __shared__ float sh_hq[H_];
    __shared__ float sh_va[H_];
    __shared__ float red[N_][4];
    __shared__ float sw[N_];
    __shared__ float ssum;

    for (int i = tid; i < H_; i += 256) { sh_hq[i] = hq[(size_t)b * H_ + i]; sh_va[i] = Va[i]; }
    __syncthreads();

    int n = tid >> 2, part = tid & 3;
    {
        const float* ep = enc_proj + ((size_t)b * N_ + n) * H_;
        float s = 0.0f;
        int h0 = part * (H_ / 4);
        for (int h = h0; h < h0 + H_ / 4; ++h) {
            float v = sh_hq[h] + ep[h];
            v = v > 0.0f ? v : 0.0f;
            s += v * sh_va[h];
        }
        red[n][part] = s;
    }
    __syncthreads();
    if (part == 0) sw[n] = red[n][0] + red[n][1] + red[n][2] + red[n][3] + bVa[0];
    __syncthreads();
    if (tid == 0) {
        float m = sw[0];
        for (int i = 1; i < N_; ++i) m = fmaxf(m, sw[i]);
        float sum = 0.0f;
        for (int i = 0; i < N_; ++i) { float e = __expf(sw[i] - m); sw[i] = e; sum += e; }
        ssum = sum;
    }
    __syncthreads();
    if (tid < N_) {
        float w = sw[tid] / ssum;
        sw[tid] = w;
        attn_out[(size_t)b * attn_ld + tid] = w;
    }
    __syncthreads();

    // gated context -> bf16 inp[:, E_:E_+F_]
    for (int f = tid; f < F_; f += 256) {
        float acc = 0.0f;
        for (int nn = 0; nn < N_; ++nn)
            acc += feat[((size_t)b * N_ + nn) * F_ + f] * sw[nn];
        acc *= gate[(size_t)b * F_ + f];
        inp_bf[(size_t)b * (E_ + F_) + E_ + f] = f2bf(acc);
    }
    // x -> bf16 inp[:, 0:E_]
    for (int e = tid; e < E_; e += 256)
        inp_bf[(size_t)b * (E_ + F_) + e] = f2bf(x[(size_t)b * E_ + e]);
}

// LSTM pointwise (pytorch gate order i, f, g, o), updates h/c in place,
// mirrors h into bf16 for the next GEMMs.
__global__ void lstm_pointwise(const float* __restrict__ gates,
                               float* __restrict__ h, float* __restrict__ c,
                               unsigned short* __restrict__ h_bf) {
    int idx = blockIdx.x * blockDim.x + threadIdx.x;
    if (idx >= B_ * H_) return;
    int b = idx >> 9, hh = idx & (H_ - 1);
    const float* g = gates + (size_t)b * 4 * H_;
    float gi = sigmf(g[hh]);
    float gf = sigmf(g[H_ + hh]);
    float gg = tanhf(g[2 * H_ + hh]);
    float go = sigmf(g[3 * H_ + hh]);
    float c2 = gf * c[idx] + gi * gg;
    float h2 = go * tanhf(c2);
    c[idx] = c2;
    h[idx] = h2;
    h_bf[idx] = f2bf(h2);
}

// Greedy argmax over V per batch row + gather next embedding into x.
__global__ __launch_bounds__(256) void argmax_gather(
        const float* __restrict__ out, int ld,
        const float* __restrict__ emb, float* __restrict__ x) {
    int b = blockIdx.x, tid = threadIdx.x;
    const float* row = out + (size_t)b * ld;
    float best = -INFINITY; int bi = 0x7FFFFFFF;
    for (int v = tid; v < V_; v += 256) {
        float val = row[v];
        if (val > best) { best = val; bi = v; }
    }
    __shared__ float sv[256];
    __shared__ int   si[256];
    sv[tid] = best; si[tid] = bi;
    __syncthreads();
    for (int s = 128; s > 0; s >>= 1) {
        if (tid < s) {
            if (sv[tid + s] > sv[tid] ||
                (sv[tid + s] == sv[tid] && si[tid + s] < si[tid])) {
                sv[tid] = sv[tid + s]; si[tid] = si[tid + s];
            }
        }
        __syncthreads();
    }
    int idx = si[0];
    for (int e = tid; e < E_; e += 256)
        x[(size_t)b * E_ + e] = emb[(size_t)idx * E_ + e];
}

__global__ void copy_hc(const float* __restrict__ h, const float* __restrict__ c,
                        float* __restrict__ dh, float* __restrict__ dc) {
    int i = blockIdx.x * blockDim.x + threadIdx.x;
    if (i < B_ * H_) { dh[i] = h[i]; dc[i] = c[i]; }
}

// ---------------------------------------------------------------------------
extern "C" void kernel_launch(void* const* d_in, const int* in_sizes, int n_in,
                              void* d_out, int out_size, void* d_ws, size_t ws_size,
                              hipStream_t stream) {
    const float* feat  = (const float*)d_in[0];
    // d_in[1] = caption (unused by greedy decode), d_in[2] = max_caption (==T_)
    const float* emb   = (const float*)d_in[3];
    const float* W1    = (const float*)d_in[4];
    const float* b1    = (const float*)d_in[5];
    const float* W2    = (const float*)d_in[6];
    const float* b2    = (const float*)d_in[7];
    const float* Va    = (const float*)d_in[8];
    const float* bVa   = (const float*)d_in[9];
    const float* Wh    = (const float*)d_in[10];
    const float* bh    = (const float*)d_in[11];
    const float* Wc    = (const float*)d_in[12];
    const float* bc    = (const float*)d_in[13];
    const float* Wg    = (const float*)d_in[14];
    const float* bg    = (const float*)d_in[15];
    const float* Wx    = (const float*)d_in[16];
    const float* Whh   = (const float*)d_in[17];
    const float* blstm = (const float*)d_in[18];
    const float* Wout  = (const float*)d_in[19];
    const float* bout  = (const float*)d_in[20];

    // Output layout: decoder_outputs [B,T,V], h [B,H], c [B,H], attentions [B,T,N]
    float* outs     = (float*)d_out;
    float* h_out    = outs + (size_t)B_ * T_ * V_;
    float* c_out    = h_out + B_ * H_;
    float* attn_out = c_out + B_ * H_;

    char* ws = (char*)d_ws;
    size_t off = 0;
    auto alloc = [&](size_t bytes) -> char* {
        char* p = ws + off;
        off = (off + bytes + 255) & ~(size_t)255;
        return p;
    };

    unsigned short* featbf = (unsigned short*)alloc((size_t)B_ * N_ * F_ * 2);
    unsigned short* W2t    = (unsigned short*)alloc((size_t)H_ * F_ * 2);        // [H][F]
    unsigned short* Wht    = (unsigned short*)alloc((size_t)H_ * F_ * 2);
    unsigned short* Wct    = (unsigned short*)alloc((size_t)H_ * F_ * 2);
    unsigned short* W1t    = (unsigned short*)alloc((size_t)H_ * H_ * 2);
    unsigned short* Wgt    = (unsigned short*)alloc((size_t)F_ * H_ * 2);        // [F][H]
    unsigned short* Wxt    = (unsigned short*)alloc((size_t)4 * H_ * (E_ + F_) * 2);
    unsigned short* Whht   = (unsigned short*)alloc((size_t)4 * H_ * H_ * 2);
    unsigned short* Woutt  = (unsigned short*)alloc((size_t)V_ * H_ * 2);
    unsigned short* meanbf = (unsigned short*)alloc((size_t)B_ * F_ * 2);
    unsigned short* hbf    = (unsigned short*)alloc((size_t)B_ * H_ * 2);
    unsigned short* inpbf  = (unsigned short*)alloc((size_t)B_ * (E_ + F_) * 2);
    float* enc_proj = (float*)alloc((size_t)B_ * N_ * H_ * 4);
    float* mean_enc = (float*)alloc((size_t)B_ * F_ * 4);
    float* h        = (float*)alloc((size_t)B_ * H_ * 4);
    float* c        = (float*)alloc((size_t)B_ * H_ * 4);
    float* hq       = (float*)alloc((size_t)B_ * H_ * 4);
    float* gate     = (float*)alloc((size_t)B_ * F_ * 4);
    float* gates    = (float*)alloc((size_t)B_ * 4 * H_ * 4);
    float* x        = (float*)alloc((size_t)B_ * E_ * 4);

    // ---- one-time conversions (weights stay resident in 192MB L2 afterwards)
    cvt_kernel<<<(B_ * N_ * F_ + 255) / 256, 256, 0, stream>>>(feat, featbf, B_ * N_ * F_);
    transpose_cvt<<<(F_ * H_ + 255) / 256, 256, 0, stream>>>(W2, W2t, F_, H_);
    transpose_cvt<<<(F_ * H_ + 255) / 256, 256, 0, stream>>>(Wh, Wht, F_, H_);
    transpose_cvt<<<(F_ * H_ + 255) / 256, 256, 0, stream>>>(Wc, Wct, F_, H_);
    transpose_cvt<<<(H_ * H_ + 255) / 256, 256, 0, stream>>>(W1, W1t, H_, H_);
    transpose_cvt<<<(H_ * F_ + 255) / 256, 256, 0, stream>>>(Wg, Wgt, H_, F_);
    transpose_cvt<<<((E_ + F_) * 4 * H_ + 255) / 256, 256, 0, stream>>>(Wx, Wxt, E_ + F_, 4 * H_);
    transpose_cvt<<<(H_ * 4 * H_ + 255) / 256, 256, 0, stream>>>(Whh, Whht, H_, 4 * H_);
    transpose_cvt<<<(H_ * V_ + 255) / 256, 256, 0, stream>>>(Wout, Woutt, H_, V_);
    mean_kernel<<<B_, 256, 0, stream>>>(feat, mean_enc, meanbf);
    init_x<<<B_, 256, 0, stream>>>(emb, x);

    auto gemm = [&](const unsigned short* A1, int K1,
                    const unsigned short* A2, int K2,
                    const unsigned short* Bt1, const unsigned short* Bt2,
                    const float* bias, float* C, int ldc,
                    unsigned short* Cbf, int M, int N, int act) {
        int ntn = N / 16;
        int groups = (M / 16) * ((ntn + 3) / 4);
        wmma_gemm<<<(groups + 3) / 4, 128, 0, stream>>>(A1, K1, A2, K2, Bt1, Bt2,
                                                        bias, C, ldc, Cbf, M, N, act);
    };

    // enc_proj = feat @ W2 + b2   (M = B*N = 2048)
    gemm(featbf, F_, nullptr, 0, W2t, nullptr, b2, enc_proj, H_, nullptr, B_ * N_, H_, 0);
    // h0 = mean @ Wh + bh (also mirror to bf16); c0 = mean @ Wc + bc
    gemm(meanbf, F_, nullptr, 0, Wht, nullptr, bh, h, H_, hbf, B_, H_, 0);
    gemm(meanbf, F_, nullptr, 0, Wct, nullptr, bc, c, H_, nullptr, B_, H_, 0);

    for (int t = 0; t < T_; ++t) {
        // hq = h @ W1 + b1
        gemm(hbf, H_, nullptr, 0, W1t, nullptr, b1, hq, H_, nullptr, B_, H_, 0);
        // gate = sigmoid(h @ Wg + bg)
        gemm(hbf, H_, nullptr, 0, Wgt, nullptr, bg, gate, F_, nullptr, B_, F_, 1);
        // attention: softmax scores, gated context, build bf16 inp = [x | ctx]
        attention_kernel<<<B_, 256, 0, stream>>>(hq, enc_proj, Va, bVa, feat, gate, x,
                                                 attn_out + (size_t)t * N_, T_ * N_, inpbf);
        // gates = inp @ Wx + h @ Whh + blstm   (fused dual-A WMMA GEMM)
        gemm(inpbf, E_ + F_, hbf, H_, Wxt, Whht, blstm, gates, 4 * H_, nullptr, B_, 4 * H_, 0);
        // LSTM cell pointwise -> h2, c2 (+ bf16 mirror of h2)
        lstm_pointwise<<<(B_ * H_ + 255) / 256, 256, 0, stream>>>(gates, h, c, hbf);
        // out = h2 @ Wout + bout, written straight into d_out[b, t, :] (ldc = T*V)
        gemm(hbf, H_, nullptr, 0, Woutt, nullptr, bout,
             outs + (size_t)t * V_, T_ * V_, nullptr, B_, V_, 0);
        // greedy next token + embedding gather into x
        argmax_gather<<<B_, 256, 0, stream>>>(outs + (size_t)t * V_, T_ * V_, emb, x);
    }

    copy_hc<<<(B_ * H_ + 255) / 256, 256, 0, stream>>>(h, c, h_out, c_out);

    (void)in_sizes; (void)n_in; (void)out_size; (void)ws_size;
}
